// MANNLanguageModel_74268574482524
// MI455X (gfx1250) — matrix operand order repaired and checked by reading, not compiled, not run on hardware
//
#include <hip/hip_runtime.h>

typedef __attribute__((ext_vector_type(16))) _Float16 v16h;
typedef __attribute__((ext_vector_type(8)))  float    v8f;

#define DEV static __device__ __forceinline__

constexpr int V = 32000, E = 256, H = 512, NSLOT = 256, D = 128;
constexpr int B = 8, S = 512;
constexpr int G4 = 4 * H;      // 2048
constexpr int HD = H + D;      // 640

DEV int lane_id() { return threadIdx.x & 31; }
DEV int wave_id() { return threadIdx.x >> 5; }

// ---------------- CDNA5 async copy to LDS (ASYNCcnt path) -------------------
DEV void async_b128(const void* g, void* l) {
  unsigned ldsoff = (unsigned)(unsigned long long)l;   // LDS aperture: low 32 bits
  asm volatile("global_load_async_to_lds_b128 %0, %1, off"
               :: "v"(ldsoff), "v"(g) : "memory");
}
DEV void wait_async0() { asm volatile("s_wait_asynccnt 0" ::: "memory"); }

// ---------------- WMMA fragment helpers (A/B contiguous k = 16*half + e; C/D per CDNA5 ISA 7.12.2) -------
// A 16x32 (f16), row-major source, rows beyond `rows` are zero
DEV v16h ldA_h16(const _Float16* src, int lda, int lane, int rows) {
  int hf = lane >> 4, r = lane & 15;
  v16h a;
  if (r < rows) {
    const _Float16* p = src + r * lda;
#pragma unroll
    for (int e = 0; e < 16; ++e) { int k = 16 * hf + e; a[e] = p[k]; }
  } else {
#pragma unroll
    for (int e = 0; e < 16; ++e) a[e] = (_Float16)0.f;
  }
  return a;
}
// A 16x32 from f32 source (convert)
DEV v16h ldA_f32(const float* src, int lda, int lane, int rows) {
  int hf = lane >> 4, r = lane & 15;
  v16h a;
  if (r < rows) {
    const float* p = src + r * lda;
#pragma unroll
    for (int e = 0; e < 16; ++e) { int k = 16 * hf + e; a[e] = (_Float16)p[k]; }
  } else {
#pragma unroll
    for (int e = 0; e < 16; ++e) a[e] = (_Float16)0.f;
  }
  return a;
}
// A 16x32 with transposed f32 source: A[m][k] = src[k][m], k >= kvalid is zero
DEV v16h ldA_f32T(const float* src, int lda, int lane, int kvalid) {
  int hf = lane >> 4, r = lane & 15;
  v16h a;
#pragma unroll
  for (int e = 0; e < 16; ++e) {
    int k = 16 * hf + e;
    a[e] = (k < kvalid) ? (_Float16)src[k * lda + r] : (_Float16)0.f;
  }
  return a;
}
// B 32x16 where B[k][n] = Wt[n][k]; Wt f16 row-major pre-offset to (n0,k0); 32B aligned
DEV v16h ldB_h16T(const _Float16* Wt, int ldw, int lane) {
  return *(const v16h*)(Wt + (lane & 15) * ldw + (lane >> 4) * 16);
}
// B 32x16 where B[k][n] = Wt[n][k]; Wt f32 row-major pre-offset to (n0,k0)
DEV v16h ldB_f32T(const float* Wt, int ldw, int lane) {
  const float* p = Wt + (lane & 15) * ldw + (lane >> 4) * 16;
  v16h b;
#pragma unroll
  for (int e = 0; e < 16; ++e) b[e] = (_Float16)p[e];
  return b;
}
// B 32x16 where B[k][n] = Sm[k][n]; f32 row-major pre-offset to (k0,n0); k >= kvalid zero
DEV v16h ldB_f32(const float* Sm, int lds_, int lane, int kvalid) {
  int n = lane & 15, hf = lane >> 4;
  v16h b;
#pragma unroll
  for (int e = 0; e < 16; ++e) {
    int k = 16 * hf + e;
    b[e] = (k < kvalid) ? (_Float16)Sm[k * lds_ + n] : (_Float16)0.f;
  }
  return b;
}
DEV v8f ldC_f32(const float* C, long ldc, int lane, int rows) {
  int n = lane & 15, hf = lane >> 4;
  v8f c;
#pragma unroll
  for (int v = 0; v < 8; ++v) { int m = v + 8 * hf; c[v] = (m < rows) ? C[(long)m * ldc + n] : 0.f; }
  return c;
}
DEV void stD_f32(float* Dst, long ldc, int lane, int rows, v8f c) {
  int n = lane & 15, hf = lane >> 4;
#pragma unroll
  for (int v = 0; v < 8; ++v) { int m = v + 8 * hf; if (m < rows) Dst[(long)m * ldc + n] = c[v]; }
}
DEV v8f wmma16(v16h a, v16h b, v8f c) {
  return __builtin_amdgcn_wmma_f32_16x16x32_f16(false, a, false, b, (short)0, c, false, false);
}
DEV float wsum(float v)  { for (int o = 16; o; o >>= 1) v += __shfl_xor(v, o, 32); return v; }
DEV float wmaxr(float v) { for (int o = 16; o; o >>= 1) v = fmaxf(v, __shfl_xor(v, o, 32)); return v; }
DEV float sigm(float x)  { return 1.f / (1.f + __expf(-x)); }

// ---------------- prep kernels ----------------------------------------------
__global__ void k_cvt_h(const float* s, _Float16* d, int n) {
  int i = blockIdx.x * 256 + threadIdx.x;
  if (i < n) d[i] = (_Float16)s[i];
}
__global__ void k_gather(const int* ids, const float* emb, _Float16* Xin) {
  long r = blockIdx.x;
  Xin[r * E + threadIdx.x] = (_Float16)emb[(long)ids[r] * E + threadIdx.x];
}

// ---------------- pre-gates: preG = Xin @ Wih^T + (b_ih + b_hh) -------------
__global__ void k_pregates(const _Float16* Xin, const _Float16* WihH,
                           const float* bih, const float* bhh, float* preG) {
  __shared__ _Float16 As[64 * E];                       // 32 KB
  int m0 = blockIdx.x * 64, nB = blockIdx.y * 128;
  for (int i = threadIdx.x; i < (64 * E) / 8; i += 256)
    async_b128(Xin + (long)m0 * E + (long)i * 8, &As[i * 8]);
  wait_async0();
  __syncthreads();
  int w = wave_id(), lane = lane_id();
  int mi = w & 3, ni = w >> 2;                          // 4 x 2 waves
  v8f acc[4];
#pragma unroll
  for (int j = 0; j < 4; ++j) {
    float bv = bih[nB + ni * 64 + j * 16 + (lane & 15)] + bhh[nB + ni * 64 + j * 16 + (lane & 15)];
#pragma unroll
    for (int v = 0; v < 8; ++v) acc[j][v] = bv;
  }
  for (int kt = 0; kt < E / 32; ++kt) {
    v16h a = ldA_h16(&As[mi * 16 * E + kt * 32], E, lane, 16);
#pragma unroll
    for (int j = 0; j < 4; ++j) {
      long n0 = nB + ni * 64 + j * 16;
      acc[j] = wmma16(a, ldB_h16T(WihH + n0 * E + kt * 32, E, lane), acc[j]);
    }
  }
#pragma unroll
  for (int j = 0; j < 4; ++j) {
    long n0 = nB + ni * 64 + j * 16;
    stD_f32(preG + (long)(m0 + mi * 16) * G4 + n0, G4, lane, 16, acc[j]);
  }
}

// ---------------- LSTM recurrence (persistent single workgroup) -------------
__global__ void k_lstm(const float* preG, const _Float16* WhhH, float* comb) {
  __shared__ _Float16 hb[16 * H];                       // 16 KB, rows 8..15 zero
  __shared__ float gates[B * G4];                       // 64 KB
  int tid = threadIdx.x, lane = lane_id(), w = wave_id();
  for (int i = tid; i < 16 * H; i += 256) hb[i] = (_Float16)0.f;
  float creg[16];
#pragma unroll
  for (int j = 0; j < 16; ++j) creg[j] = 0.f;
  __syncthreads();
  for (int t = 0; t < S; ++t) {
    // 128 N-tiles over 8 waves, two tiles in flight per wave (2x WMMA ILP)
    for (int nt = 0; nt < 8; ++nt) {
      int n0 = w * 256 + nt * 32;
      int n1 = n0 + 16;
      v8f acc0 = ldC_f32(preG + (long)t * G4 + n0, (long)S * G4, lane, B);
      v8f acc1 = ldC_f32(preG + (long)t * G4 + n1, (long)S * G4, lane, B);
      for (int kt = 0; kt < H / 32; ++kt) {
        v16h a = ldA_h16(&hb[kt * 32], H, lane, 16);
        acc0 = wmma16(a, ldB_h16T(WhhH + (long)n0 * H + kt * 32, H, lane), acc0);
        acc1 = wmma16(a, ldB_h16T(WhhH + (long)n1 * H + kt * 32, H, lane), acc1);
      }
      stD_f32(&gates[n0], G4, lane, B, acc0);
      stD_f32(&gates[n1], G4, lane, B, acc1);
    }
    __syncthreads();
#pragma unroll
    for (int j = 0; j < 16; ++j) {                      // 4096 (b,h) pairs
      int idx = tid + 256 * j;
      int b = idx >> 9, h = idx & 511;
      float gi = sigm(gates[b * G4 + h]);
      float gf = sigm(gates[b * G4 + 512 + h]);
      float gg = tanhf(gates[b * G4 + 1024 + h]);
      float go = sigm(gates[b * G4 + 1536 + h]);
      float c = gf * creg[j] + gi * gg;
      creg[j] = c;
      float hh = go * tanhf(c);
      hb[b * H + h] = (_Float16)hh;
      comb[((long)t * B + b) * HD + h] = hh;
    }
    __syncthreads();
  }
}

// ---------------- NTM memory scan (M resident in LDS) -----------------------
__global__ void k_mem(const float* M0, float* comb,
                      const float* Wr, const float* br, const float* Ww, const float* bw,
                      const float* Wks, const float* bks,
                      const float* We, const float* be, const float* Wa, const float* ba) {
  __shared__ float Mem[NSLOT * D];                      // 128 KB
  __shared__ float hb[B * H];                           // 16 KB
  __shared__ float proj[B * 512];                       // 16 KB: rq|wq|er_raw|ad_raw
  __shared__ float er[B * D], ad[B * D];                // 8 KB (pre-scaled 1/B)
  __shared__ float sc[2 * B * NSLOT];                   // 16 KB
  __shared__ float Mn[NSLOT];
  __shared__ float qn[2 * B], beta[B];
  int tid = threadIdx.x, lane = lane_id(), w = wave_id();
  for (int i = tid; i < NSLOT * D; i += 256) Mem[i] = M0[i];
  int pb = w >> 1;
  const float* Wsel = (pb == 0) ? Wr : (pb == 1) ? Ww : (pb == 2) ? We : Wa;
  const float* bsel = (pb == 0) ? br : (pb == 1) ? bw : (pb == 2) ? be : ba;
  __syncthreads();
  for (int t = 0; t < S; ++t) {
    for (int i = tid; i < B * H; i += 256)
      hb[i] = comb[((long)t * B + (i >> 9)) * HD + (i & 511)];
    __syncthreads();
    // projections: proj[:, pb*128 + c] = h @ Wsel^T + bsel
#pragma unroll
    for (int j = 0; j < 4; ++j) {
      int c0 = ((w & 1) * 4 + j) * 16;
      v8f acc;
      float bv = bsel[c0 + (lane & 15)];
#pragma unroll
      for (int v = 0; v < 8; ++v) acc[v] = bv;
      for (int kt = 0; kt < H / 32; ++kt) {
        v16h a = ldA_f32(&hb[kt * 32], H, lane, B);
        acc = wmma16(a, ldB_f32T(Wsel + (long)c0 * H + kt * 32, H, lane), acc);
      }
      stD_f32(&proj[pb * 128 + c0], 512, lane, B, acc);
    }
    __syncthreads();
    // elementwise: erase/add (scaled 1/B), beta, q-norms, M row norms
    for (int i = tid; i < B * D; i += 256) {
      int b = i >> 7, d = i & 127;
      er[i] = sigm(proj[b * 512 + 256 + d]) * (1.f / B);
      ad[i] = tanhf(proj[b * 512 + 384 + d]) * (1.f / B);
    }
    {
      int b = w;
      float s = 0.f, s0 = 0.f, s1 = 0.f;
      for (int j = lane; j < H; j += 32) s += hb[b * H + j] * Wks[j];
      for (int j = lane; j < D; j += 32) {
        float x0 = proj[b * 512 + j], x1 = proj[b * 512 + 128 + j];
        s0 += x0 * x0; s1 += x1 * x1;
      }
      s = wsum(s); s0 = wsum(s0); s1 = wsum(s1);
      if (lane == 0) {
        float x = s + bks[0];
        beta[b] = (x > 20.f) ? x : log1pf(__expf(x));
        qn[b]     = fmaxf(sqrtf(s0), 1e-8f);
        qn[B + b] = fmaxf(sqrtf(s1), 1e-8f);
      }
    }
    {
      float s = 0.f;
      for (int j = 0; j < D; ++j) { int d = (j + lane) & (D - 1); float x = Mem[tid * D + d]; s += x * x; }
      Mn[tid] = fmaxf(sqrtf(s), 1e-8f);
    }
    __syncthreads();
    // cosine scores (beta-scaled for read) : 32 tiles over 8 waves
#pragma unroll
    for (int j = 0; j < 4; ++j) {
      int tile = w * 4 + j, q = tile >> 4, n0 = (tile & 15) * 16;
      v8f acc = {};
      for (int kt = 0; kt < D / 32; ++kt) {
        v16h a = ldA_f32(&proj[q * 128 + kt * 32], 512, lane, B);
        acc = wmma16(a, ldB_f32T(&Mem[(long)n0 * D + kt * 32], D, lane), acc);
      }
      int nn = n0 + (lane & 15);
#pragma unroll
      for (int v = 0; v < 8; ++v) {
        int m = v + 8 * (lane >> 4);
        if (m < B) {
          float val = acc[v] / (qn[q * B + m] * Mn[nn]);
          if (q == 0) val *= beta[m];
          sc[(q * B + m) * NSLOT + nn] = val;
        }
      }
    }
    __syncthreads();
    // softmax rows: wave w -> batch w, both q
    for (int q = 0; q < 2; ++q) {
      float* row = &sc[(q * B + w) * NSLOT];
      float mx = -3.4e38f;
      for (int j = lane; j < NSLOT; j += 32) mx = fmaxf(mx, row[j]);
      mx = wmaxr(mx);
      float sm = 0.f;
      for (int j = lane; j < NSLOT; j += 32) { float e2 = __expf(row[j] - mx); row[j] = e2; sm += e2; }
      sm = wsum(sm);
      float inv = 1.f / sm;
      for (int j = lane; j < NSLOT; j += 32) row[j] *= inv;
    }
    __syncthreads();
    // read_vec = rw @ M -> comb[t][b][512 + d]
    {
      int n0 = w * 16;
      v8f acc = {};
      for (int kt = 0; kt < NSLOT / 32; ++kt) {
        v16h a = ldA_f32(&sc[kt * 32], NSLOT, lane, B);
        acc = wmma16(a, ldB_f32(&Mem[(long)kt * 32 * D + n0], D, lane, 32), acc);
      }
#pragma unroll
      for (int v = 0; v < 8; ++v) {
        int m = v + 8 * (lane >> 4);
        if (m < B) comb[((long)t * B + m) * HD + H + n0 + (lane & 15)] = acc[v];
      }
    }
    __syncthreads();
    // M = M*(1 - ww^T er) + ww^T ad : 128 tiles over 8 waves, rank-8 (K padded)
#pragma unroll
    for (int j = 0; j < 16; ++j) {
      int tile = w * 16 + j;
      int m0 = (tile >> 3) * 16, n0 = (tile & 7) * 16;
      v16h aT = ldA_f32T(&sc[B * NSLOT + m0], NSLOT, lane, B);
      v8f accE = {}, accA = {};
      accE = wmma16(aT, ldB_f32(&er[n0], D, lane, B), accE);
      accA = wmma16(aT, ldB_f32(&ad[n0], D, lane, B), accA);
#pragma unroll
      for (int v = 0; v < 8; ++v) {
        int m = v + 8 * (lane >> 4);
        float* p = &Mem[(m0 + m) * D + n0 + (lane & 15)];
        *p = *p * (1.f - accE[v]) + accA[v];
      }
    }
    __syncthreads();
  }
}

// ---------------- LayerNorm + transpose to [B,S,HD] f16 ---------------------
__global__ void k_ln(const float* comb, const float* g, const float* bb, _Float16* Xn) {
  int row = blockIdx.x * 8 + wave_id();                 // comb row = t*B + b
  int lane = lane_id();
  const float* x = comb + (long)row * HD;
  float s = 0.f;
  for (int j = lane; j < HD; j += 32) s += x[j];
  float mu = wsum(s) / HD;
  float v2 = 0.f;
  for (int j = lane; j < HD; j += 32) { float d = x[j] - mu; v2 += d * d; }
  float inv = rsqrtf(wsum(v2) / HD + 1e-5f);
  int t = row >> 3, b = row & 7;
  _Float16* y = Xn + ((long)(b * S + t)) * HD;
  for (int j = lane; j < HD; j += 32) y[j] = (_Float16)((x[j] - mu) * inv * g[j] + bb[j]);
}

// ---------------- output projection: logits = Xn @ Wout^T + b_out -----------
__global__ void k_out(const _Float16* Xn, const _Float16* WoH, const float* bo, float* logits) {
  __shared__ _Float16 As[64 * HD];                      // 80 KB
  int m0 = blockIdx.x * 64, nB = blockIdx.y * 128;
  // async DMA the A tile (64 x 640 f16 = 80 KB) into LDS, no VGPR round-trip
  for (int i = threadIdx.x; i < (64 * HD) / 8; i += 256)
    async_b128(Xn + (long)m0 * HD + (long)i * 8, &As[i * 8]);
  wait_async0();
  __syncthreads();
  int w = wave_id(), lane = lane_id();
  int mi = w & 3, ni = w >> 2;
  v8f acc[4];
#pragma unroll
  for (int j = 0; j < 4; ++j) {
    float bv = bo[nB + ni * 64 + j * 16 + (lane & 15)];
#pragma unroll
    for (int v = 0; v < 8; ++v) acc[j][v] = bv;
  }
  // software-pipelined K loop: B frags for kt+1 stream while kt's WMMAs issue
  v16h bcur[4];
#pragma unroll
  for (int j = 0; j < 4; ++j)
    bcur[j] = ldB_h16T(WoH + (long)(nB + ni * 64 + j * 16) * HD, HD, lane);
  for (int kt = 0; kt < HD / 32; ++kt) {                // 20 K-steps
    v16h a = ldA_h16(&As[mi * 16 * HD + kt * 32], HD, lane, 16);
    v16h bnext[4];
    if (kt + 1 < HD / 32) {
#pragma unroll
      for (int j = 0; j < 4; ++j) {
        const _Float16* wpt = WoH + (long)(nB + ni * 64 + j * 16) * HD + (kt + 1) * 32;
        __builtin_prefetch(wpt + 32, 0, 1);             // next K-slab hint
        bnext[j] = ldB_h16T(wpt, HD, lane);
      }
    }
#pragma unroll
    for (int j = 0; j < 4; ++j) acc[j] = wmma16(a, bcur[j], acc[j]);
    if (kt + 1 < HD / 32) {
#pragma unroll
      for (int j = 0; j < 4; ++j) bcur[j] = bnext[j];
    }
  }
#pragma unroll
  for (int j = 0; j < 4; ++j) {
    long n0 = nB + ni * 64 + j * 16;
    stD_f32(logits + (long)(m0 + mi * 16) * V + n0, V, lane, 16, acc[j]);
  }
}

// ---------------- host ------------------------------------------------------
extern "C" void kernel_launch(void* const* d_in, const int* in_sizes, int n_in,
                              void* d_out, int out_size, void* d_ws, size_t ws_size,
                              hipStream_t stream) {
  const int*   ids = (const int*)  d_in[0];
  const float* emb = (const float*)d_in[1];
  const float* wih = (const float*)d_in[2];
  const float* whh = (const float*)d_in[3];
  const float* bih = (const float*)d_in[4];
  const float* bhh = (const float*)d_in[5];
  const float* M0  = (const float*)d_in[6];
  const float* Wr  = (const float*)d_in[7];
  const float* br  = (const float*)d_in[8];
  const float* Ww  = (const float*)d_in[9];
  const float* bw  = (const float*)d_in[10];
  const float* Wks = (const float*)d_in[11];
  const float* bks = (const float*)d_in[12];
  const float* We  = (const float*)d_in[13];
  const float* be  = (const float*)d_in[14];
  const float* Wa  = (const float*)d_in[15];
  const float* ba  = (const float*)d_in[16];
  const float* lng = (const float*)d_in[17];
  const float* lnb = (const float*)d_in[18];
  const float* Wo  = (const float*)d_in[19];
  const float* bo  = (const float*)d_in[20];
  float* logits = (float*)d_out;

  char* wp = (char*)d_ws;
  auto alloc = [&](size_t bytes) { void* p = wp; wp += (bytes + 255) & ~size_t(255); return p; };
  _Float16* Xin  = (_Float16*)alloc((size_t)B * S * E * 2);
  _Float16* wihH = (_Float16*)alloc((size_t)G4 * E * 2);
  _Float16* whhH = (_Float16*)alloc((size_t)G4 * H * 2);
  _Float16* woH  = (_Float16*)alloc((size_t)V * HD * 2);
  float*    preG = (float*)   alloc((size_t)B * S * G4 * 4);
  float*    comb = (float*)   alloc((size_t)S * B * HD * 4);
  _Float16* Xn   = (_Float16*)alloc((size_t)B * S * HD * 2);

  k_cvt_h<<<(G4 * E + 255) / 256, 256, 0, stream>>>(wih, wihH, G4 * E);
  k_cvt_h<<<(G4 * H + 255) / 256, 256, 0, stream>>>(whh, whhH, G4 * H);
  k_cvt_h<<<(V * HD + 255) / 256, 256, 0, stream>>>(Wo, woH, V * HD);
  k_gather<<<B * S, 256, 0, stream>>>(ids, emb, Xin);
  k_pregates<<<dim3(B * S / 64, G4 / 128), 256, 0, stream>>>(Xin, wihH, bih, bhh, preG);
  k_lstm<<<1, 256, 0, stream>>>(preG, whhH, comb);
  k_mem<<<1, 256, 0, stream>>>(M0, comb, Wr, br, Ww, bw, Wks, bks, We, be, Wa, ba);
  k_ln<<<(B * S) / 8, 256, 0, stream>>>(comb, lng, lnb, Xn);
  k_out<<<dim3(B * S / 64, V / 128), 256, 0, stream>>>(Xn, woH, bo, logits);
}
